// GNN_12678743458219
// MI455X (gfx1250) — matrix operand order repaired and checked by reading, not compiled, not run on hardware
//
#include <hip/hip_runtime.h>
#include <hip/hip_bf16.h>
#include <hip/hip_fp16.h>

// GNN energy model: R=4,B=256,N=200,NF=5,H=50,OUT=2
// Memory-bound on adj (164MB @ 23.3TB/s ~ 7us). Dense 200x200x50 matmul per
// (r,b) done with v_wmma_f32_16x16x32_f16: adjacency 0/1 is exact in f16,
// accumulation in f32. One workgroup (8 waves) per batch element, r-loop
// inside => deterministic. LDS = 161,888 B -> 2 workgroups per 320KB WGP.

#define R_ 4
#define B_ 256
#define N_ 200
#define NF_ 5
#define H_ 50

#define KP 232            // K pitch (halfs): 16B-aligned rows, 116-dword
                          // stride => conflict-free ds_load_b128 across lanes
#define HP 56             // h row pitch (halfs)
#define MT 13             // ceil(200/16) M tiles
#define NT 4              // ceil(50/16)  N tiles (features)
#define KTILES 7          // ceil(200/32) K steps of 32

typedef __attribute__((ext_vector_type(16))) _Float16 v16h;
typedef __attribute__((ext_vector_type(8)))  _Float16 v8h;
typedef __attribute__((ext_vector_type(4)))  _Float16 v4h;
typedef __attribute__((ext_vector_type(8)))  float    v8f;

// LDS carve-up (bytes)
#define OFF_A    0                      // A f16 [208][KP]      96512
#define OFF_QT   96512                  // QT f16 [64][KP]      29696
#define OFF_H    126208                 // h f16 [208][HP]      23296
#define OFF_UT   149504                 // uT f16 [16][KP]       7424
#define OFF_UF   156928                 // uF f32 [2][208]       1664
#define OFF_DI   158592                 // dinv f32 [208]         832
#define OFF_EA   159424                 // eatm f32 [208]         832
#define OFF_PRM  160256                 // params f32 [405]      1632
#define SMEM_BYTES 161888

static __device__ __forceinline__ v16h cat16(v8h lo, v8h hi) {
  v16h r;
#pragma unroll
  for (int i = 0; i < 8; ++i) { r[i] = lo[i]; r[i + 8] = hi[i]; }
  return r;
}

static __device__ __forceinline__ float silu(float s) {
  return s / (1.0f + __expf(-s));
}

__global__ __launch_bounds__(256)
void GNN_12678743458219_kernel(const float* __restrict__ x,
                               const int*   __restrict__ adj,
                               const float* __restrict__ nu,
                               const float* __restrict__ W1,
                               const float* __restrict__ b1,
                               const float* __restrict__ W2,
                               const float* __restrict__ b2,
                               const float* __restrict__ W3,
                               const float* __restrict__ b3,
                               float* __restrict__ out) {
  extern __shared__ char smem[];
  _Float16* __restrict__ Af  = (_Float16*)(smem + OFF_A);
  _Float16* __restrict__ QT  = (_Float16*)(smem + OFF_QT);
  _Float16* __restrict__ hT  = (_Float16*)(smem + OFF_H);
  _Float16* __restrict__ uT  = (_Float16*)(smem + OFF_UT);
  float*    __restrict__ uF  = (float*)(smem + OFF_UF);
  float*    __restrict__ din = (float*)(smem + OFF_DI);
  float*    __restrict__ ea  = (float*)(smem + OFF_EA);
  float*    __restrict__ prm = (float*)(smem + OFF_PRM);
  float* pW1 = prm;        // 250
  float* pb1 = prm + 250;  // 50
  float* pW2 = prm + 300;  // 100
  float* pb2 = prm + 400;  // 2
  float* pW3 = prm + 402;  // 2
  float* pb3 = prm + 404;  // 1

  const int tid  = threadIdx.x;
  const int b    = blockIdx.x;
  const int wave = tid >> 5;
  const int lane = tid & 31;
  const int sub  = lane >> 4;   // half-wave select
  const int l15  = lane & 15;

  // ---- one-time init: params to LDS, zero pad regions ----
  for (int i = tid; i < 405; i += 256) {
    float v;
    if (i < 250)      v = W1[i];
    else if (i < 300) v = b1[i - 250];
    else if (i < 400) v = W2[i - 300];
    else if (i < 402) v = b2[i - 400];
    else if (i < 404) v = W3[i - 402];
    else              v = b3[0];
    prm[i] = v;
  }
  // zero K-pad columns of A (cols 200..231, all 208 rows)
  for (int i = tid; i < 208 * 32; i += 256) {
    int rr = i >> 5, cc = 200 + (i & 31);
    Af[rr * KP + cc] = (_Float16)0.0f;
  }
  // zero M-pad rows of A (rows 200..207) so epilogues need no m-guards
  for (int rr = 200; rr < 208; ++rr)
    for (int cc = tid; cc < KP; cc += 256) Af[rr * KP + cc] = (_Float16)0.0f;
  // zero QT and uT entirely (data regions get rewritten each r; pads stay 0)
  for (int i = tid; i < 64 * KP; i += 256) QT[i] = (_Float16)0.0f;
  for (int i = tid; i < 16 * KP; i += 256) uT[i] = (_Float16)0.0f;

  float accE = 0.0f;

  for (int r = 0; r < R_; ++r) {
    __syncthreads();  // protects prev-iteration LDS reads + init

    // ---- Phase A: adj -> f16 LDS, zero eatm, prefetch next adj
    {
      const int4* adj4 = (const int4*)(adj + ((size_t)r * B_ + b) * (N_ * N_));
      for (int c = tid; c < (N_ * N_) / 4; c += 256) {  // 10000 int4 chunks
        int4 w = adj4[c];
        int e0 = c * 4;
        int row = e0 / N_;
        int col = e0 - row * N_;       // rows are 50 chunks: never crosses
        v4h hv = { (_Float16)(float)w.x, (_Float16)(float)w.y,
                   (_Float16)(float)w.z, (_Float16)(float)w.w };
        *(v4h*)(Af + row * KP + col) = hv;
      }
      for (int i = tid; i < 208; i += 256) ea[i] = 0.0f;
      if (r + 1 < R_) {  // global_prefetch_b8 of next adjacency
        const char* nx = (const char*)(adj + ((size_t)(r + 1) * B_ + b) * (N_ * N_));
        for (int off = tid * 128; off < N_ * N_ * 4; off += 256 * 128)
          __builtin_prefetch(nx + off, 0, 1);
      }
    }
    __syncthreads();

    // ---- Phase B: degrees + dinv = rsqrt(deg_adj + 1 self-loop)
    if (tid < N_) {
      const v8h* row = (const v8h*)(Af + tid * KP);
      float s = 0.0f;
#pragma unroll 5
      for (int c = 0; c < N_ / 8; ++c) {
        v8h h8 = row[c];
#pragma unroll
        for (int i = 0; i < 8; ++i) s += (float)h8[i];
      }
      din[tid] = rsqrtf(s + 1.0f);
    }
    __syncthreads();

    // ---- Phase C: QT[f][n] = dinv[n] * (x @ W1)[n][f]   (f16, K-major)
    if (tid < N_) {
      const int n = tid;
      const float* xp = x + ((size_t)r * B_ + b) * (N_ * NF_) + n * NF_;
      float xv[NF_];
#pragma unroll
      for (int c = 0; c < NF_; ++c) xv[c] = xp[c];
      const float di = din[n];
      for (int f = 0; f < H_; ++f) {
        float p = 0.0f;
#pragma unroll
        for (int c = 0; c < NF_; ++c) p += xv[c] * pW1[c * H_ + f];
        QT[f * KP + n] = (_Float16)(di * p);
      }
    }
    __syncthreads();

    // ---- Phase D: S = A @ Q via WMMA; h = silu(dinv*S + b1) -> f16 LDS
    // nt = wave & 3 is loop-invariant => compiler keeps all 7 B frags in VGPRs
    for (int t = wave; t < MT * NT; t += 8) {
      const int mt = t >> 2, nt = t & 3;
      const _Float16* Arow = Af + (mt * 16 + l15) * KP;
      const _Float16* Brow = QT + (nt * 16 + l15) * KP;
      v8f acc = {};
      // software-pipelined A-fragment loads around the WMMA chain
      v8h alo = *(const v8h*)(Arow + sub * 8);
      v8h ahi = *(const v8h*)(Arow + 16 + sub * 8);
#pragma unroll
      for (int kt = 0; kt < KTILES; ++kt) {
        const int k0 = kt * 32;
        v8h blo = *(const v8h*)(Brow + k0 + sub * 8);
        v8h bhi = *(const v8h*)(Brow + k0 + 16 + sub * 8);
        v8h nlo = alo, nhi = ahi;
        if (kt + 1 < KTILES) {
          nlo = *(const v8h*)(Arow + k0 + 32 + sub * 8);
          nhi = *(const v8h*)(Arow + k0 + 48 + sub * 8);
        }
        acc = __builtin_amdgcn_wmma_f32_16x16x32_f16(
            false, cat16(alo, ahi), false, cat16(blo, bhi),
            (short)0, acc, false, false);
        alo = nlo; ahi = nhi;
      }
      const int f = nt * 16 + l15;
      if (f < H_) {
        const float bf = pb1[f];
#pragma unroll
        for (int v = 0; v < 8; ++v) {
          const int m = mt * 16 + v + sub * 8;  // C/D layout: VGPR v -> M
          float s = din[m] * acc[v] + bf;       // rows 200..207 are zero-fed
          hT[m * HP + f] = (_Float16)silu(s);   // and never read back
        }
      }
    }
    __syncthreads();

    // ---- Phase E: u = dinv * (h @ W2)  (200x2), staged f16 K-major
    for (int idx = tid; idx < N_ * 2; idx += 256) {
      const int m = idx >> 1, j = idx & 1;
      float s = 0.0f;
#pragma unroll 10
      for (int f = 0; f < H_; ++f) s += (float)hT[m * HP + f] * pW2[f * 2 + j];
      const float u = din[m] * s;
      uF[j * 208 + m] = u;
      uT[j * KP + m]  = (_Float16)u;
    }
    __syncthreads();

    // ---- Phase F: V = A @ u via WMMA; per-atom energy contributions
    for (int mt = wave; mt < MT; mt += 8) {
      const _Float16* Arow = Af + (mt * 16 + l15) * KP;
      const _Float16* Brow = uT + l15 * KP;
      v8f acc = {};
      v8h alo = *(const v8h*)(Arow + sub * 8);
      v8h ahi = *(const v8h*)(Arow + 16 + sub * 8);
#pragma unroll
      for (int kt = 0; kt < KTILES; ++kt) {
        const int k0 = kt * 32;
        v8h blo = *(const v8h*)(Brow + k0 + sub * 8);
        v8h bhi = *(const v8h*)(Brow + k0 + 16 + sub * 8);
        v8h nlo = alo, nhi = ahi;
        if (kt + 1 < KTILES) {
          nlo = *(const v8h*)(Arow + k0 + 32 + sub * 8);
          nhi = *(const v8h*)(Arow + k0 + 48 + sub * 8);
        }
        acc = __builtin_amdgcn_wmma_f32_16x16x32_f16(
            false, cat16(alo, ahi), false, cat16(blo, bhi),
            (short)0, acc, false, false);
        alo = nlo; ahi = nhi;
      }
      const int j = l15;
      if (j < 2) {
        const float w3 = pW3[j], b2j = pb2[j];
#pragma unroll
        for (int v = 0; v < 8; ++v) {
          const int m = mt * 16 + v + sub * 8;
          const float h2 = din[m] * (acc[v] + uF[j * 208 + m]) + b2j;
          atomicAdd(&ea[m], w3 * h2);   // ds_add_f32; rows>=200 never read
        }
      }
    }
    __syncthreads();

    // ---- Phase G: wave-0 tree reduction over atoms, apply nu[b][r]
    if (wave == 0) {
      float e = 0.0f;
      for (int m = lane; m < N_; m += 32) e += ea[m];
#pragma unroll
      for (int off = 16; off > 0; off >>= 1) e += __shfl_down(e, off, 32);
      if (lane == 0) {
        e += (float)N_ * pb3[0];          // + b3 per atom
        accE += e * nu[b * R_ + r];       // nu is [B,R]
      }
    }
  }

  if (tid == 0) out[b] = accE;
}

extern "C" void kernel_launch(void* const* d_in, const int* in_sizes, int n_in,
                              void* d_out, int out_size, void* d_ws, size_t ws_size,
                              hipStream_t stream) {
  const float* x   = (const float*)d_in[0];
  const int*   adj = (const int*)d_in[1];
  const float* nu  = (const float*)d_in[2];
  const float* W1  = (const float*)d_in[3];
  const float* b1  = (const float*)d_in[4];
  const float* W2  = (const float*)d_in[5];
  const float* b2  = (const float*)d_in[6];
  const float* W3  = (const float*)d_in[7];
  const float* b3  = (const float*)d_in[8];
  float* out = (float*)d_out;
  GNN_12678743458219_kernel<<<dim3(B_), dim3(256), SMEM_BYTES, stream>>>(
      x, adj, nu, W1, b1, W2, b2, W3, b3, out);
  (void)in_sizes; (void)n_in; (void)out_size; (void)d_ws; (void)ws_size;
}
